// Distiller_10591389352047
// MI455X (gfx1250) — matrix operand order repaired and checked by reading, not compiled, not run on hardware
//
#include <hip/hip_runtime.h>
#include <hip/hip_bf16.h>
#include <math.h>

// ---------------------------------------------------------------------------
// Types for CDNA5 WMMA (wave32, 16x16x32 bf16 -> f32)
// ---------------------------------------------------------------------------
typedef __bf16 bf16_t;
typedef __attribute__((ext_vector_type(16))) __bf16 v16bf;
typedef __attribute__((ext_vector_type(8)))  __bf16 v8bf;
typedef __attribute__((ext_vector_type(8)))  float  v8f;

#define B_    4
#define CT_   512
#define CS_   320
#define HF_   48
#define M_    2304          // 48*48 tokens
#define C_    21
#define HW_   9216          // 96*96
#define NH_   8
#define HD_   40
#define HDP_  64            // head dim padded to K multiple of 32
#define DFF_  2048
#define KIM_  2880          // 320*9 conv im2col K

#define OUT_SOUT_N   (B_*C_*HW_)   // 774144
#define OUT_IC_IDX   (OUT_SOUT_N + 0)
#define OUT_SA_IDX   (OUT_SOUT_N + 1)
#define OUT_PI_IDX   (OUT_SOUT_N + 2)

// ---------------------------------------------------------------------------
// gfx1250 async global->LDS path (ASYNCcnt), with synchronous fallback.
// Probe round 2 confirmed: builtin exists; param 0 is non-const AS1 (global)
// pointer to int __attribute__((vector_size(16))); param 1 is the LDS side.
// ---------------------------------------------------------------------------
#if defined(__has_builtin)
#if __has_builtin(__builtin_amdgcn_global_load_async_to_lds_b128) && \
    __has_builtin(__builtin_amdgcn_s_wait_asynccnt)
#define HAVE_ASYNC_LDS 1
#endif
#endif

#ifdef HAVE_ASYNC_LDS
typedef int async_b128_t __attribute__((vector_size(16)));
typedef __attribute__((address_space(1))) async_b128_t* g_b128_p;  // global
typedef __attribute__((address_space(3))) async_b128_t* l_b128_p;  // LDS
#endif

__device__ __forceinline__ void cp16_g2l(const bf16_t* __restrict__ g,
                                         bf16_t* __restrict__ l) {
#ifdef HAVE_ASYNC_LDS
  __builtin_amdgcn_global_load_async_to_lds_b128((g_b128_p)g, (l_b128_p)l,
                                                 /*offset=*/0, /*cpol=*/0);
#else
  *(uint4*)l = *(const uint4*)g;
#endif
}
__device__ __forceinline__ void wait_async_lds() {
#ifdef HAVE_ASYNC_LDS
  __builtin_amdgcn_s_wait_asynccnt(0);
#endif
}

// ---------------------------------------------------------------------------
// wave32 reductions
// ---------------------------------------------------------------------------
__device__ __forceinline__ float warp_sum(float v) {
#pragma unroll
  for (int o = 16; o > 0; o >>= 1) v += __shfl_xor(v, o, 32);
  return v;
}
__device__ __forceinline__ float warp_max(float v) {
#pragma unroll
  for (int o = 16; o > 0; o >>= 1) v = fmaxf(v, __shfl_xor(v, o, 32));
  return v;
}
__device__ __forceinline__ float warp_min(float v) {
#pragma unroll
  for (int o = 16; o > 0; o >>= 1) v = fminf(v, __shfl_xor(v, o, 32));
  return v;
}

// ---------------------------------------------------------------------------
// LDS-staged, double-buffered bf16 WMMA GEMM.
//   C[m,n] = act( scale * sum_k A[m,k]*Bt[n,k] + bias[n] + addsrc[m,n] )
// Block: 256 threads (8 waves, 2x4), tile 128(M) x 64(N), K-step 32.
// Wave computes 64x16 (4 WMMAs/K-step, B fragment reused 4x).
// Requirements: M % 128 == 0, Ndim % 64 == 0, K % 32 == 0,
//               lda/ldb multiples of 8 (16B aligned rows). Stores clamp to Nout.
// ---------------------------------------------------------------------------
#define BM_ 128
#define BN_ 64
#define BK_ 32

__global__ void __launch_bounds__(256)
gemm_bf16_nt(const bf16_t* __restrict__ A, int lda,
             const bf16_t* __restrict__ Bt, int ldb,
             int Mdim, int Ndim, int Nout, int Kdim, float scale,
             const float* __restrict__ bias,
             const float* __restrict__ addsrc, int ldadd,
             int relu, int store_cm,
             float* __restrict__ Cf, bf16_t* __restrict__ Cb, int ldc)
{
  __shared__ __align__(32) bf16_t As[2][BM_][BK_];   // 2 x 8 KB
  __shared__ __align__(32) bf16_t Bs[2][BN_][BK_];   // 2 x 4 KB

  const int tid  = threadIdx.x;
  const int lane = tid & 31;
  const int wid  = tid >> 5;
  const int hi   = lane >> 4;     // half-wave select
  const int lrow = lane & 15;
  const int wr   = wid >> 2;      // 0..1 -> M offset 64*wr
  const int wc   = wid & 3;       // 0..3 -> N offset 16*wc
  const size_t mBase = (size_t)blockIdx.x * BM_;
  const size_t nBase = (size_t)blockIdx.y * BN_;

  const bf16_t* Ag = A  + mBase * lda;
  const bf16_t* Bg = Bt + nBase * ldb;

  // Stage one 128x32 A tile + 64x32 B tile into LDS buffer `buf`.
  auto stage = [&](int buf, int k0) {
    // A: 512 x 16B chunks; this thread does chunks tid and tid+256.
    {
      int r = tid >> 2, kc = tid & 3;
      cp16_g2l(Ag + (size_t)r * lda + k0 + kc * 8, &As[buf][r][kc * 8]);
      int c1 = tid + 256;
      r = c1 >> 2; kc = c1 & 3;
      cp16_g2l(Ag + (size_t)r * lda + k0 + kc * 8, &As[buf][r][kc * 8]);
    }
    // B: 256 x 16B chunks; one per thread.
    {
      int r = tid >> 2, kc = tid & 3;
      cp16_g2l(Bg + (size_t)r * ldb + k0 + kc * 8, &Bs[buf][r][kc * 8]);
    }
  };

  v8f acc[4] = {{}, {}, {}, {}};
  const int nk = Kdim / BK_;

  stage(0, 0);
  for (int i = 0; i < nk; ++i) {
    wait_async_lds();        // this wave's stage done (ASYNCcnt -> 0)
    __syncthreads();         // tile i resident block-wide; tile i-1 consumed
    if (i + 1 < nk) stage((i + 1) & 1, (i + 1) * BK_);
    const int buf = i & 1;

    // B fragment: lane holds Bt row (n), 16 contiguous K at offset 16*hi.
    v16bf bv = *(const v16bf*)&Bs[buf][wc * 16 + lrow][16 * hi];
#pragma unroll
    for (int r4 = 0; r4 < 4; ++r4) {
      const bf16_t* ar = &As[buf][wr * 64 + r4 * 16 + lrow][0];
      v8bf alo = *(const v8bf*)(ar + 16 * hi);       // K [16*hi, 16*hi+8)
      v8bf ahi = *(const v8bf*)(ar + 16 * hi + 8);   // K [16*hi+8, 16*hi+16)
      v16bf av;
#pragma unroll
      for (int e = 0; e < 8; ++e) { av[e] = alo[e]; av[e + 8] = ahi[e]; }
      acc[r4] = __builtin_amdgcn_wmma_f32_16x16x32_bf16(
          /*neg_a=*/false, av, /*neg_b=*/false, bv,
          /*c_mod=*/(short)0, acc[r4], /*reuse_a=*/false, /*reuse_b=*/false);
    }
  }

  const int n = (int)nBase + wc * 16 + lrow;
  if (n >= Nout) return;
  const float bv0 = bias ? bias[n] : 0.0f;
#pragma unroll
  for (int r4 = 0; r4 < 4; ++r4) {
#pragma unroll
    for (int r = 0; r < 8; ++r) {
      int m = (int)mBase + wr * 64 + r4 * 16 + r + 8 * hi;  // C: VGPR r -> M=r+8*hi
      float v = acc[r4][r] * scale + bv0;
      if (addsrc) v += addsrc[(size_t)m * ldadd + n];
      if (relu)   v = fmaxf(v, 0.0f);
      size_t idx = store_cm ? ((size_t)n * ldc + m) : ((size_t)m * ldc + n);
      if (Cf) Cf[idx] = v;
      if (Cb) Cb[idx] = (bf16_t)v;
    }
  }
}

// ---------------------------------------------------------------------------
// Elementwise helpers
// ---------------------------------------------------------------------------
__global__ void cast_f32_bf16_k(const float* __restrict__ in,
                                bf16_t* __restrict__ out, size_t n) {
  size_t i = (size_t)blockIdx.x * 256 + threadIdx.x;
  if (i < n) out[i] = (bf16_t)in[i];
}

// TFt[b][c*M + m] = t_feat[b][m*CT + c]  (bf16 transpose for X@TF B-operand)
__global__ void tft_cast_k(const float* __restrict__ tf, bf16_t* __restrict__ tft) {
  size_t i = (size_t)blockIdx.x * 256 + threadIdx.x;
  if (i >= (size_t)B_ * M_ * CT_) return;
  int c = (int)(i % CT_);
  size_t t = i / CT_;
  int m = (int)(t % M_);
  int b = (int)(t / M_);
  tft[((size_t)b * CT_ + c) * M_ + m] = (bf16_t)tf[i];
}

// Split fused qkv (bf16, ld=960) into padded per-head Q/K (M x 64) and V^T (64 x M)
__global__ void qkv_split_k(const bf16_t* __restrict__ qkv,
                            bf16_t* __restrict__ Qp, bf16_t* __restrict__ Kp,
                            bf16_t* __restrict__ Vt) {
  size_t i = (size_t)blockIdx.x * 256 + threadIdx.x;
  if (i >= (size_t)B_ * NH_ * M_ * HDP_) return;
  int e = (int)(i % HDP_);
  size_t t = i / HDP_;
  int m = (int)(t % M_);
  t /= M_;
  int h = (int)(t % NH_);
  int b = (int)(t / NH_);
  const bf16_t* src = qkv + ((size_t)b * M_ + m) * (3 * CS_);
  bf16_t z = (bf16_t)0.0f;
  bf16_t q = (e < HD_) ? src[h * HD_ + e] : z;
  bf16_t k = (e < HD_) ? src[CS_ + h * HD_ + e] : z;
  size_t qi = (((size_t)(b * NH_ + h)) * M_ + m) * HDP_ + e;
  Qp[qi] = q;
  Kp[qi] = k;
  bf16_t v = (e < HD_) ? src[2 * CS_ + h * HD_ + e] : z;
  Vt[((size_t)(b * NH_ + h) * HDP_ + e) * M_ + m] = v;   // padded to 64 rows
}

// Row softmax: fp32 scores row -> bf16 probs row. One block per row.
__global__ void __launch_bounds__(256)
softmax_rows_k(const float* __restrict__ S, bf16_t* __restrict__ P, int L) {
  __shared__ float sred[8];
  __shared__ float sval;
  const int t = threadIdx.x, lane = t & 31, wv = t >> 5;
  const float* x = S + (size_t)blockIdx.x * L;
  float mx = -1e30f;
  for (int i = t; i < L; i += 256) mx = fmaxf(mx, x[i]);
  mx = warp_max(mx);
  if (lane == 0) sred[wv] = mx;
  __syncthreads();
  if (t == 0) { float m = sred[0]; for (int i = 1; i < 8; ++i) m = fmaxf(m, sred[i]); sval = m; }
  __syncthreads();
  mx = sval;
  float s = 0.0f;
  for (int i = t; i < L; i += 256) s += __expf(x[i] - mx);
  s = warp_sum(s);
  if (lane == 0) sred[wv] = s;
  __syncthreads();
  if (t == 0) { float ss = 0; for (int i = 0; i < 8; ++i) ss += sred[i]; sval = ss; }
  __syncthreads();
  const float inv = 1.0f / sval;
  bf16_t* prow = P + (size_t)blockIdx.x * L;
  for (int i = t; i < L; i += 256) prow[i] = (bf16_t)(__expf(x[i] - mx) * inv);
}

// LayerNorm of (x + add) over d, one wave per row; writes f32 and/or bf16.
__global__ void __launch_bounds__(256)
ln_residual_k(const float* __restrict__ x, const float* __restrict__ add,
              const float* __restrict__ g, const float* __restrict__ bb,
              float* __restrict__ outf, bf16_t* __restrict__ outb,
              int rows, int d) {
  int row = blockIdx.x * 8 + (threadIdx.x >> 5);
  if (row >= rows) return;
  int lane = threadIdx.x & 31;
  const float* xr = x   + (size_t)row * d;
  const float* ar = add + (size_t)row * d;
  float s = 0.0f, s2 = 0.0f;
  for (int i = lane; i < d; i += 32) {
    float v = xr[i] + ar[i];
    s += v; s2 += v * v;
  }
  s = warp_sum(s); s2 = warp_sum(s2);
  float mean = s / d;
  float var  = s2 / d - mean * mean;
  float inv  = rsqrtf(var + 1e-5f);
  for (int i = lane; i < d; i += 32) {
    float v = ((xr[i] + ar[i]) - mean) * inv * g[i] + bb[i];
    if (outf) outf[(size_t)row * d + i] = v;
    if (outb) outb[(size_t)row * d + i] = (bf16_t)v;
  }
}

// im2col for 3x3 pad-1 conv over one batch: enc_b is flat (CS, 48, 48) view.
__global__ void im2col_k(const float* __restrict__ enc_b, bf16_t* __restrict__ A2) {
  size_t i = (size_t)blockIdx.x * 256 + threadIdx.x;
  if (i >= (size_t)M_ * KIM_) return;
  int col = (int)(i % KIM_);
  int p   = (int)(i / KIM_);
  int ci = col / 9, r = col % 9;
  int ky = r / 3, kx = r % 3;
  int y = p / HF_, xx = p % HF_;
  int yy = y + ky - 1, xc = xx + kx - 1;
  float v = (yy >= 0 && yy < HF_ && xc >= 0 && xc < HF_)
                ? enc_b[(size_t)ci * M_ + yy * HF_ + xc] : 0.0f;
  A2[i] = (bf16_t)v;
}

// BN train-mode stats per output channel over (B, H, W).
__global__ void __launch_bounds__(256)
bn_stats_k(const float* __restrict__ O, float* __restrict__ bnmean,
           float* __restrict__ bnvar) {
  int co = blockIdx.x;
  float s = 0.0f, s2 = 0.0f;
  for (int i = threadIdx.x; i < B_ * M_; i += 256) {
    int b = i / M_, p = i - b * M_;
    float v = O[((size_t)b * CT_ + co) * M_ + p];
    s += v; s2 += v * v;
  }
  s = warp_sum(s); s2 = warp_sum(s2);
  __shared__ float ra[8], rb[8];
  int lane = threadIdx.x & 31, wv = threadIdx.x >> 5;
  if (lane == 0) { ra[wv] = s; rb[wv] = s2; }
  __syncthreads();
  if (threadIdx.x == 0) {
    float ts = 0, ts2 = 0;
    for (int i = 0; i < 8; ++i) { ts += ra[i]; ts2 += rb[i]; }
    float m = ts / (B_ * M_);
    bnmean[co] = m;
    bnvar[co]  = ts2 / (B_ * M_) - m * m;
  }
}

// SA loss: per (b, j) column (stride M over 512 channels) of the scrambled
// (cT, M) views of G and conv_bn output; L2-normalize both, accumulate ||diff||.
__global__ void __launch_bounds__(256)
sa_loss_k(const float* __restrict__ G, const float* __restrict__ O,
          const float* __restrict__ mean, const float* __restrict__ var,
          const float* __restrict__ bng, const float* __restrict__ bnb,
          float* __restrict__ out_sa) {
  int col = blockIdx.x * 8 + (threadIdx.x >> 5);
  if (col >= B_ * M_) return;
  int b = col / M_, j = col - b * M_;
  int lane = threadIdx.x & 31;
  const float* Gb = G + (size_t)b * M_ * CT_;
  const float* Ob = O + (size_t)b * CT_ * M_;
  float sg = 0.0f, sf = 0.0f;
  for (int i = lane; i < CT_; i += 32) {
    float g = Gb[(size_t)i * M_ + j];
    float f = (Ob[(size_t)i * M_ + j] - mean[i]) * rsqrtf(var[i] + 1e-5f) * bng[i] + bnb[i];
    sg += g * g; sf += f * f;
  }
  sg = warp_sum(sg); sf = warp_sum(sf);
  float ig  = 1.0f / fmaxf(sqrtf(sg), 1e-12f);
  float iff = 1.0f / fmaxf(sqrtf(sf), 1e-12f);
  float sd = 0.0f;
  for (int i = lane; i < CT_; i += 32) {
    float g = Gb[(size_t)i * M_ + j] * ig;
    float f = ((Ob[(size_t)i * M_ + j] - mean[i]) * rsqrtf(var[i] + 1e-5f) * bng[i] + bnb[i]) * iff;
    float d = g - f;
    sd += d * d;
  }
  sd = warp_sum(sd);
  if (lane == 0) atomicAdd(out_sa, sqrtf(sd) * (1.0f / (float)(M_ * B_)));
}

// KL(pi) loss: one thread per pixel, channel stride HW.
__global__ void __launch_bounds__(256)
pi_loss_k(const float* __restrict__ t_out, const float* __restrict__ s_out,
          float* __restrict__ out_pi) {
  int idx = blockIdx.x * 256 + threadIdx.x;
  float acc = 0.0f;
  if (idx < B_ * HW_) {
    int b = idx / HW_, p = idx - b * HW_;
    const float* t = t_out + (size_t)b * C_ * HW_ + p;
    const float* s = s_out + (size_t)b * C_ * HW_ + p;
    float mt = -1e30f, ms = -1e30f;
    for (int c = 0; c < C_; ++c) {
      mt = fmaxf(mt, t[(size_t)c * HW_]);
      ms = fmaxf(ms, s[(size_t)c * HW_]);
    }
    float st = 0.0f, ss = 0.0f;
    for (int c = 0; c < C_; ++c) {
      st += __expf(t[(size_t)c * HW_] - mt);
      ss += __expf(s[(size_t)c * HW_] - ms);
    }
    float lt = __logf(st), ls = __logf(ss);
    for (int c = 0; c < C_; ++c) {
      float lpt = t[(size_t)c * HW_] - mt - lt;
      float lps = s[(size_t)c * HW_] - ms - ls;
      acc += __expf(lpt) * (lpt - lps);
    }
  }
  acc = warp_sum(acc);
  if ((threadIdx.x & 31) == 0 && acc != 0.0f)
    atomicAdd(out_pi, acc * (1.0f / (float)(B_ * C_ * HW_)));
}

// IC loss support: per-batch teacher logit max/min.
__global__ void __launch_bounds__(256)
ic_minmax_k(const float* __restrict__ t_out, float* __restrict__ mm) {
  int b = blockIdx.x;
  const float* t = t_out + (size_t)b * C_ * HW_;
  float mx = -1e30f, mn = 1e30f;
  for (int i = threadIdx.x; i < C_ * HW_; i += 256) {
    float v = t[i];
    mx = fmaxf(mx, v); mn = fminf(mn, v);
  }
  mx = warp_max(mx); mn = warp_min(mn);
  __shared__ float ra[8], rb[8];
  int lane = threadIdx.x & 31, wv = threadIdx.x >> 5;
  if (lane == 0) { ra[wv] = mx; rb[wv] = mn; }
  __syncthreads();
  if (threadIdx.x == 0) {
    float a = ra[0], c = rb[0];
    for (int i = 1; i < 8; ++i) { a = fmaxf(a, ra[i]); c = fminf(c, rb[i]); }
    mm[b * 2] = a; mm[b * 2 + 1] = c;
  }
}

// Masked teacher-logit correction -> ct (b, C, HW) fp32.
__global__ void ic_correct_k(const float* __restrict__ t_out,
                             const int* __restrict__ y,
                             const float* __restrict__ mm,
                             float* __restrict__ ct) {
  int idx = blockIdx.x * 256 + threadIdx.x;
  if (idx >= B_ * HW_) return;
  int b = idx / HW_, p = idx - b * HW_;
  const float* t = t_out + (size_t)b * C_ * HW_ + p;
  int yy = y[(size_t)b * HW_ + p];
  if (yy == 255) yy = 0;
  int pred = 0; float best = t[0];
  for (int c = 1; c < C_; ++c) {
    float v = t[(size_t)c * HW_];
    if (v > best) { best = v; pred = c; }
  }
  bool mask = (yy != pred);
  float vmx = mm[b * 2], vmn = mm[b * 2 + 1];
  for (int c = 0; c < C_; ++c) {
    float v = t[(size_t)c * HW_];
    if (mask) v = (c == yy) ? vmx : vmn;
    ct[((size_t)b * C_ + c) * HW_ + p] = v;
  }
}

// Per-batch 21x21 Grams, row-normalize, accumulate squared diff.
__global__ void __launch_bounds__(256)
ic_gram_k(const float* __restrict__ ct, const float* __restrict__ s_out,
          float* __restrict__ out_ic) {
  int b = blockIdx.x;
  __shared__ float gT[C_ * C_], gS[C_ * C_], nT[C_], nS[C_];
  const float* T = ct    + (size_t)b * C_ * HW_;
  const float* S = s_out + (size_t)b * C_ * HW_;
  for (int idx = threadIdx.x; idx < C_ * C_; idx += 256) {
    int i = idx / C_, j = idx % C_;
    const float* ti = T + (size_t)i * HW_;
    const float* tj = T + (size_t)j * HW_;
    const float* si = S + (size_t)i * HW_;
    const float* sj = S + (size_t)j * HW_;
    float at = 0.0f, as = 0.0f;
    for (int m = 0; m < HW_; ++m) { at += ti[m] * tj[m]; as += si[m] * sj[m]; }
    gT[idx] = at; gS[idx] = as;
  }
  __syncthreads();
  for (int i = threadIdx.x; i < C_; i += 256) {
    float st = 0.0f, ss = 0.0f;
    for (int j = 0; j < C_; ++j) {
      st += gT[i * C_ + j] * gT[i * C_ + j];
      ss += gS[i * C_ + j] * gS[i * C_ + j];
    }
    nT[i] = 1.0f / fmaxf(sqrtf(st), 1e-12f);
    nS[i] = 1.0f / fmaxf(sqrtf(ss), 1e-12f);
  }
  __syncthreads();
  float acc = 0.0f;
  for (int idx = threadIdx.x; idx < C_ * C_; idx += 256) {
    int i = idx / C_;
    float d = gS[idx] * nS[i] - gT[idx] * nT[i];
    acc += d * d;
  }
  acc = warp_sum(acc);
  __shared__ float sred[8];
  int lane = threadIdx.x & 31, wv = threadIdx.x >> 5;
  if (lane == 0) sred[wv] = acc;
  __syncthreads();
  if (threadIdx.x == 0) {
    float s = 0.0f;
    for (int i = 0; i < 8; ++i) s += sred[i];
    atomicAdd(out_ic, s * (1.0f / (float)(C_ * B_)));
  }
}

__global__ void copy_f32_k(const float* __restrict__ in, float* __restrict__ out,
                           size_t n) {
  size_t i = (size_t)blockIdx.x * 256 + threadIdx.x;
  if (i < n) out[i] = in[i];
}
__global__ void zero_f32_k(float* __restrict__ out, int n) {
  int i = blockIdx.x * 256 + threadIdx.x;
  if (i < n) out[i] = 0.0f;
}

// ---------------------------------------------------------------------------
// Host-side launch plumbing
// ---------------------------------------------------------------------------
static inline void launch_gemm(hipStream_t s, const bf16_t* A, int lda,
                               const bf16_t* Bt, int ldb, int M, int N, int Nout,
                               int K, float scale, const float* bias,
                               const float* add, int ldadd, int relu, int cm,
                               float* Cf, bf16_t* Cb, int ldc) {
  dim3 g(M / BM_, N / BN_);
  gemm_bf16_nt<<<g, dim3(256), 0, s>>>(A, lda, Bt, ldb, M, N, Nout, K, scale,
                                       bias, add, ldadd, relu, cm, Cf, Cb, ldc);
}
static inline unsigned blks(size_t n) { return (unsigned)((n + 255) / 256); }

extern "C" void kernel_launch(void* const* d_in, const int* in_sizes, int n_in,
                              void* d_out, int out_size, void* d_ws, size_t ws_size,
                              hipStream_t stream) {
  (void)in_sizes; (void)n_in; (void)out_size; (void)ws_size;
  const float* t_feat = (const float*)d_in[0];
  const float* s_feat = (const float*)d_in[1];
  const float* t_out  = (const float*)d_in[2];
  const float* s_out  = (const float*)d_in[3];
  const int*   y      = (const int*)d_in[4];
  const float* conv_w = (const float*)d_in[5];
  const float* bn_g   = (const float*)d_in[6];
  const float* bn_b   = (const float*)d_in[7];
  const float* inw    = (const float*)d_in[8];
  const float* inb    = (const float*)d_in[9];
  const float* outw   = (const float*)d_in[10];
  const float* outb   = (const float*)d_in[11];
  const float* ln1g   = (const float*)d_in[12];
  const float* ln1b   = (const float*)d_in[13];
  const float* ln2g   = (const float*)d_in[14];
  const float* ln2b   = (const float*)d_in[15];
  const float* ff1w   = (const float*)d_in[16];
  const float* ff1b   = (const float*)d_in[17];
  const float* ff2w   = (const float*)d_in[18];
  const float* ff2b   = (const float*)d_in[19];
  float* out = (float*)d_out;

  // Workspace bump allocator (256B aligned).
  char* base = (char*)d_ws;
  size_t off = 0;
  auto alloc = [&](size_t bytes) -> void* {
    void* p = base + off;
    off += (bytes + 255) & ~(size_t)255;
    return p;
  };
  bf16_t* tf_bf   = (bf16_t*)alloc((size_t)B_ * M_ * CT_ * 2);
  bf16_t* tft_bf  = (bf16_t*)alloc((size_t)B_ * CT_ * M_ * 2);
  bf16_t* x0_bf   = (bf16_t*)alloc((size_t)B_ * M_ * CS_ * 2);
  bf16_t* inw_bf  = (bf16_t*)alloc((size_t)3 * CS_ * CS_ * 2);
  bf16_t* outw_bf = (bf16_t*)alloc((size_t)CS_ * CS_ * 2);
  bf16_t* ff1w_bf = (bf16_t*)alloc((size_t)DFF_ * CS_ * 2);
  bf16_t* ff2w_bf = (bf16_t*)alloc((size_t)CS_ * DFF_ * 2);
  bf16_t* cw_bf   = (bf16_t*)alloc((size_t)CT_ * KIM_ * 2);
  float*  Sbuf    = (float*) alloc((size_t)M_ * M_ * 4);      // reused scores
  bf16_t* Pbuf    = (bf16_t*)alloc((size_t)M_ * M_ * 2);      // reused probs
  float*  G       = (float*) alloc((size_t)B_ * M_ * CT_ * 4);
  bf16_t* qkv_bf  = (bf16_t*)alloc((size_t)B_ * M_ * 3 * CS_ * 2);
  bf16_t* Qp      = (bf16_t*)alloc((size_t)B_ * NH_ * M_ * HDP_ * 2);
  bf16_t* Kp      = (bf16_t*)alloc((size_t)B_ * NH_ * M_ * HDP_ * 2);
  bf16_t* Vt      = (bf16_t*)alloc((size_t)B_ * NH_ * HDP_ * M_ * 2);
  bf16_t* o_bf    = (bf16_t*)alloc((size_t)B_ * M_ * CS_ * 2);
  float*  attn    = (float*) alloc((size_t)B_ * M_ * CS_ * 4);
  float*  x1f     = (float*) alloc((size_t)B_ * M_ * CS_ * 4);
  bf16_t* x1b     = (bf16_t*)alloc((size_t)B_ * M_ * CS_ * 2);
  bf16_t* h1      = (bf16_t*)alloc((size_t)B_ * M_ * DFF_ * 2);
  float*  f2      = (float*) alloc((size_t)B_ * M_ * CS_ * 4);
  float*  enc     = (float*) alloc((size_t)B_ * M_ * CS_ * 4);
  bf16_t* im2c    = (bf16_t*)alloc((size_t)M_ * KIM_ * 2);    // per-batch reuse
  float*  O       = (float*) alloc((size_t)B_ * CT_ * M_ * 4);
  float*  bnmean  = (float*) alloc(CT_ * 4);
  float*  bnvar   = (float*) alloc(CT_ * 4);
  float*  mm      = (float*) alloc(B_ * 2 * 4);
  float*  ctb     = (float*) alloc((size_t)B_ * C_ * HW_ * 4);

  // 0) Output init: copy s_out, zero the three loss scalars.
  copy_f32_k<<<blks(OUT_SOUT_N), 256, 0, stream>>>(s_out, out, (size_t)OUT_SOUT_N);
  zero_f32_k<<<1, 256, 0, stream>>>(out + OUT_SOUT_N, 3);

  // 1) Precision staging (f32 -> bf16 operands for WMMA).
  cast_f32_bf16_k<<<blks((size_t)B_ * M_ * CT_), 256, 0, stream>>>(t_feat, tf_bf, (size_t)B_ * M_ * CT_);
  tft_cast_k<<<blks((size_t)B_ * M_ * CT_), 256, 0, stream>>>(t_feat, tft_bf);
  cast_f32_bf16_k<<<blks((size_t)B_ * M_ * CS_), 256, 0, stream>>>(s_feat, x0_bf, (size_t)B_ * M_ * CS_);
  cast_f32_bf16_k<<<blks((size_t)3 * CS_ * CS_), 256, 0, stream>>>(inw, inw_bf, (size_t)3 * CS_ * CS_);
  cast_f32_bf16_k<<<blks((size_t)CS_ * CS_), 256, 0, stream>>>(outw, outw_bf, (size_t)CS_ * CS_);
  cast_f32_bf16_k<<<blks((size_t)DFF_ * CS_), 256, 0, stream>>>(ff1w, ff1w_bf, (size_t)DFF_ * CS_);
  cast_f32_bf16_k<<<blks((size_t)CS_ * DFF_), 256, 0, stream>>>(ff2w, ff2w_bf, (size_t)CS_ * DFF_);
  cast_f32_bf16_k<<<blks((size_t)CT_ * KIM_), 256, 0, stream>>>(conv_w, cw_bf, (size_t)CT_ * KIM_);

  // 2) SA branch: G = softmax(TF TF^T / sqrt(M)) @ TF + TF   (per batch)
  const float sa_scale = 1.0f / sqrtf((float)M_);
  for (int b = 0; b < B_; ++b) {
    const bf16_t* TFb = tf_bf + (size_t)b * M_ * CT_;
    launch_gemm(stream, TFb, CT_, TFb, CT_, M_, M_, M_, CT_, sa_scale,
                nullptr, nullptr, 0, 0, 0, Sbuf, nullptr, M_);
    softmax_rows_k<<<M_, 256, 0, stream>>>(Sbuf, Pbuf, M_);
    launch_gemm(stream, Pbuf, M_, tft_bf + (size_t)b * CT_ * M_, M_,
                M_, CT_, CT_, M_, 1.0f, nullptr,
                t_feat + (size_t)b * M_ * CT_, CT_, 0, 0,
                G + (size_t)b * M_ * CT_, nullptr, CT_);
  }

  // 3) Transformer encoder on s_feat (flat view (B*M, 320)).
  const int MT = B_ * M_;
  launch_gemm(stream, x0_bf, CS_, inw_bf, CS_, MT, 3 * CS_, 3 * CS_, CS_, 1.0f,
              inb, nullptr, 0, 0, 0, nullptr, qkv_bf, 3 * CS_);
  qkv_split_k<<<blks((size_t)B_ * NH_ * M_ * HDP_), 256, 0, stream>>>(qkv_bf, Qp, Kp, Vt);
  const float att_scale = 1.0f / sqrtf((float)HD_);
  for (int bh = 0; bh < B_ * NH_; ++bh) {
    const bf16_t* Qh = Qp + (size_t)bh * M_ * HDP_;
    const bf16_t* Kh = Kp + (size_t)bh * M_ * HDP_;
    launch_gemm(stream, Qh, HDP_, Kh, HDP_, M_, M_, M_, HDP_, att_scale,
                nullptr, nullptr, 0, 0, 0, Sbuf, nullptr, M_);
    softmax_rows_k<<<M_, 256, 0, stream>>>(Sbuf, Pbuf, M_);
    int b = bh / NH_, h = bh % NH_;
    launch_gemm(stream, Pbuf, M_, Vt + (size_t)bh * HDP_ * M_, M_,
                M_, HDP_, HD_, M_, 1.0f, nullptr, nullptr, 0, 0, 0,
                nullptr, o_bf + ((size_t)b * M_) * CS_ + h * HD_, CS_);
  }
  launch_gemm(stream, o_bf, CS_, outw_bf, CS_, MT, CS_, CS_, CS_, 1.0f,
              outb, nullptr, 0, 0, 0, attn, nullptr, CS_);
  ln_residual_k<<<(MT + 7) / 8, 256, 0, stream>>>(s_feat, attn, ln1g, ln1b,
                                                  x1f, x1b, MT, CS_);
  launch_gemm(stream, x1b, CS_, ff1w_bf, CS_, MT, DFF_, DFF_, CS_, 1.0f,
              ff1b, nullptr, 0, 1, 0, nullptr, h1, DFF_);
  launch_gemm(stream, h1, DFF_, ff2w_bf, DFF_, MT, CS_, CS_, DFF_, 1.0f,
              ff2b, nullptr, 0, 0, 0, f2, nullptr, CS_);
  ln_residual_k<<<(MT + 7) / 8, 256, 0, stream>>>(x1f, f2, ln2g, ln2b,
                                                  enc, nullptr, MT, CS_);

  // 4) 3x3 conv (implicit GEMM, K=2880) + BN stats.
  for (int b = 0; b < B_; ++b) {
    im2col_k<<<blks((size_t)M_ * KIM_), 256, 0, stream>>>(
        enc + (size_t)b * M_ * CS_, im2c);
    launch_gemm(stream, im2c, KIM_, cw_bf, KIM_, M_, CT_, CT_, KIM_, 1.0f,
                nullptr, nullptr, 0, 0, /*store_cm=*/1,
                O + (size_t)b * CT_ * M_, nullptr, M_);
  }
  bn_stats_k<<<CT_, 256, 0, stream>>>(O, bnmean, bnvar);

  // 5) SA loss (column-normalize G and BN(conv), accumulate ||diff||).
  sa_loss_k<<<(B_ * M_ + 7) / 8, 256, 0, stream>>>(G, O, bnmean, bnvar,
                                                   bn_g, bn_b, out + OUT_SA_IDX);

  // 6) pi loss (KLDiv mean).
  pi_loss_k<<<blks((size_t)B_ * HW_), 256, 0, stream>>>(t_out, s_out,
                                                        out + OUT_PI_IDX);

  // 7) ic loss.
  ic_minmax_k<<<B_, 256, 0, stream>>>(t_out, mm);
  ic_correct_k<<<blks((size_t)B_ * HW_), 256, 0, stream>>>(t_out, y, mm, ctb);
  ic_gram_k<<<B_, 256, 0, stream>>>(ctb, s_out, out + OUT_IC_IDX);
}
